// I2GCLayer_27797028339884
// MI455X (gfx1250) — matrix operand order repaired and checked by reading, not compiled, not run on hardware
//
#include <hip/hip_runtime.h>

// Problem constants (from reference): B=16, C_IN=256, C_OUT=256, N=1024
#define BATCH 16
#define CIN   256
#define COUT  256
#define NN    1024

typedef __attribute__((ext_vector_type(16))) __bf16 v16bf;
typedef __attribute__((ext_vector_type(8)))  __bf16 v8bf;
typedef __attribute__((ext_vector_type(2)))  __bf16 v2bf;
typedef __attribute__((ext_vector_type(8)))  float  v8f;
typedef __attribute__((ext_vector_type(2)))  float  v2f;

// Lowered by clang to native v_cvt_pk_bf16_f32 (RNE) on gfx1250.
__device__ __forceinline__ __bf16 cvt_bf16(float f) { return (__bf16)f; }

// ---------------------------------------------------------------------------
// Kernel 1: Y[b][o][n] = sum_c W[o][c] * X[b][c][n], stored bf16 in workspace.
// One 16x16 tile of Y per wave; K-loop over c in steps of 32 with
// v_wmma_f32_16x16x32_bf16 (f32 accumulate). ~2.1 GFLOP total, W is 256 KB
// (L2-resident), so fragments are built straight from global.
// ---------------------------------------------------------------------------
__global__ __launch_bounds__(256) void i2gc_wx_kernel(
    const float* __restrict__ W, const float* __restrict__ X,
    __bf16* __restrict__ Y) {
  const int wave = blockIdx.x * 8 + (threadIdx.x >> 5);
  const int nt = wave & 63;          // 64 n-tiles
  const int ot = (wave >> 6) & 15;   // 16 o-tiles
  const int b  = wave >> 10;         // 16 batches
  const int lane = threadIdx.x & 31;
  const int r16  = lane & 15;
  const int hi   = lane >> 4;
  const int kbA  = hi * 16;          // A-frag K interleave (16-granular)
  const int kbB  = hi * 16;          // B-frag K interleave (16-granular)
  const int o0 = ot * 16, n0 = nt * 16;

  v8f acc = {};
  for (int c = 0; c < CIN; c += 32) {
    // A fragment: W rows o0..o0+15, K = c + kbA + j
    const float* wrow = W + (size_t)(o0 + r16) * CIN + c + kbA;
    v16bf a;
#pragma unroll
    for (int j = 0; j < 8; ++j)  a[j] = cvt_bf16(wrow[j]);
#pragma unroll
    for (int j = 8; j < 16; ++j) a[j] = cvt_bf16(wrow[j]);
    // B fragment: X[b][c + kbB + j][n0 + r16]
    const float* xcol = X + ((size_t)b * CIN + (c + kbB)) * NN + n0 + r16;
    v16bf bfrag;
#pragma unroll
    for (int j = 0; j < 16; ++j) bfrag[j] = cvt_bf16(xcol[(size_t)j * NN]);
    acc = __builtin_amdgcn_wmma_f32_16x16x32_bf16(
        false, a, false, bfrag, (short)0, acc, false, false);
  }
  // D layout: VGPR r -> row o0 + hi*8 + r, col n0 + r16
  __bf16* yout = Y + ((size_t)b * COUT + o0 + hi * 8) * NN + n0 + r16;
#pragma unroll
  for (int r = 0; r < 8; ++r) yout[(size_t)r * NN] = cvt_bf16(acc[r]);
}

// ---------------------------------------------------------------------------
// Kernel 2: out[b][o][m] = ReLU( sum_n Y[b][o][n]*adjH[b][n][m] + bias[o] ),
// adjH = adj1 for o<128, adj2 otherwise.
// Block = 8 waves owning a [128 o x 32 m] output slab, so each adj element is
// pulled from HBM exactly once with 128 B contiguous per-row requests.
// Cooperative fill: each thread owns a 2-col x 2-row quad of the 32x32 adj
// k-slice -> one global_load_b64 per row, one packed cvt + one aligned
// ds_store_b32 per column pair, written directly in B-fragment order.
// Double-buffered, one barrier per k-step; each lane reads each B fragment as
// one contiguous 32 B LDS access; A fragment is reused across 2 WMMAs.
// ---------------------------------------------------------------------------
__global__ __launch_bounds__(256) void i2gc_ya_kernel(
    const __bf16* __restrict__ Y, const float* __restrict__ A1,
    const float* __restrict__ A2, const float* __restrict__ bias,
    float* __restrict__ out) {
  const int mt   = blockIdx.x & 31;         // 32 m-slabs of 32 columns
  const int half = (blockIdx.x >> 5) & 1;   // which weight half / adj
  const int b    = blockIdx.x >> 6;         // 16 batches
  const int m0 = mt * 32;
  const int t = threadIdx.x;
  const int lane = t & 31;
  const int w = t >> 5;                     // wave -> o-subtile within half
  const int r16 = lane & 15;
  const int hi  = lane >> 4;
  const int kbA = hi * 16;
  const int o0 = half * 128 + w * 16;

  __shared__ __align__(32) __bf16 lds[2][2][512];  // [buf][m-subtile][lane*16+j]

  const float*  adj  = (half ? A2 : A1) + (size_t)b * NN * NN;
  const __bf16* yrow = Y + ((size_t)b * COUT + o0 + r16) * NN + kbA;

  // fill mapping: thread t owns cols (mcol, mcol+1) x rows (np0, np0+1)
  const int mcol = (t & 15) * 2;            // 0,2,..,30
  const int np0  = (t >> 4) * 2;            // 0,2,..,30
  const int mh   = mcol >> 4;               // which m-subtile
  const int l0   = (mcol & 15) + ((np0 & 16) ? 16 : 0);  // fragment lane
  const int j0   = np0 & 15;                             // fragment slot pair
  const float* fptr = adj + (size_t)np0 * NN + m0 + mcol;

  v8f acc0 = {}, acc1 = {};
  for (int nk = 0; nk < NN; nk += 32) {
    const int buf = (nk >> 5) & 1;
    {
      const float* p = fptr + (size_t)nk * NN;
      v2f r0 = *(const v2f*)p;          // row np0,   cols mcol, mcol+1
      v2f r1 = *(const v2f*)(p + NN);   // row np0+1, cols mcol, mcol+1
      v2bf p0; p0[0] = cvt_bf16(r0[0]); p0[1] = cvt_bf16(r1[0]);
      v2bf p1; p1[0] = cvt_bf16(r0[1]); p1[1] = cvt_bf16(r1[1]);
      *(v2bf*)(&lds[buf][mh][l0 * 16 + j0])       = p0;  // col mcol
      *(v2bf*)(&lds[buf][mh][(l0 + 1) * 16 + j0]) = p1;  // col mcol+1
      if (nk + 32 < NN) {  // hint next adj slice toward the caches
        __builtin_prefetch(p + (size_t)32 * NN, 0, 0);
        __builtin_prefetch(p + (size_t)33 * NN, 0, 0);
      }
    }
    __syncthreads();
    // A fragment: two contiguous 16 B bf16 chunks per lane, reused twice
    v8bf alo = *(const v8bf*)(yrow + nk);
    v8bf ahi = *(const v8bf*)(yrow + nk + 8);
    v16bf a = __builtin_shufflevector(alo, ahi, 0, 1, 2, 3, 4, 5, 6, 7, 8, 9,
                                      10, 11, 12, 13, 14, 15);
    // B fragments: one contiguous 32 B LDS read per lane per m-subtile
    v16bf b0 = *(const v16bf*)(&lds[buf][0][lane * 16]);
    v16bf b1 = *(const v16bf*)(&lds[buf][1][lane * 16]);
    acc0 = __builtin_amdgcn_wmma_f32_16x16x32_bf16(
        false, a, false, b0, (short)0, acc0, false, false);
    acc1 = __builtin_amdgcn_wmma_f32_16x16x32_bf16(
        false, a, false, b1, (short)0, acc1, false, false);
  }
  // epilogue: bias + ReLU, D layout rows o0+hi*8+r, cols m0+r16 / m0+16+r16
  float* orow = out + ((size_t)b * COUT + o0 + hi * 8) * NN + m0 + r16;
#pragma unroll
  for (int r = 0; r < 8; ++r) {
    const float bb = bias[o0 + hi * 8 + r];
    float v0 = acc0[r] + bb;
    float v1 = acc1[r] + bb;
    orow[(size_t)r * NN]      = v0 > 0.f ? v0 : 0.f;
    orow[(size_t)r * NN + 16] = v1 > 0.f ? v1 : 0.f;
  }
}

// ---------------------------------------------------------------------------
// Launch: inputs in setup_inputs() order: x, adj1, adj2, weight, bias.
// Workspace: Y bf16 [16][256][1024] = 8.39 MB.
// ---------------------------------------------------------------------------
extern "C" void kernel_launch(void* const* d_in, const int* in_sizes, int n_in,
                              void* d_out, int out_size, void* d_ws,
                              size_t ws_size, hipStream_t stream) {
  (void)in_sizes; (void)n_in; (void)out_size; (void)ws_size;
  const float* x      = (const float*)d_in[0];
  const float* adj1   = (const float*)d_in[1];
  const float* adj2   = (const float*)d_in[2];
  const float* weight = (const float*)d_in[3];
  const float* bias   = (const float*)d_in[4];
  float* out = (float*)d_out;
  __bf16* Y = (__bf16*)d_ws;

  // Kernel 1: 16*16*64 = 16384 wave-tiles / 8 waves per block
  i2gc_wx_kernel<<<2048, 256, 0, stream>>>(weight, x, Y);
  // Kernel 2: 16 batches * 2 halves * 32 m-slabs
  i2gc_ya_kernel<<<1024, 256, 0, stream>>>(Y, adj1, adj2, bias, out);
}
